// WAVESTATE_13151189860394
// MI455X (gfx1250) — compile-verified, matrix operand order checked by reading, not run on hardware
//
#include <hip/hip_runtime.h>
#include <hip/hip_bf16.h>

// ---------------------------------------------------------------------------
// WAVESTATE forward for MI455X (gfx1250, wave32, WMMA).
// All dense math runs through a batched bf16-WMMA GEMM (f32 accumulate),
// templated on operand transposes so the hot path is branch-free.
// ---------------------------------------------------------------------------

typedef unsigned short ushort_t;
typedef unsigned int   uint_t;
typedef __attribute__((ext_vector_type(16))) __bf16 v16bf;
typedef __attribute__((ext_vector_type(8)))  float  v8f;

#define DEV static __device__ __forceinline__

// ---- f32 -> bf16 pair, native fptrunc (backend picks packed cvt) ------------
DEV uint_t pk2(float a, float b) {
  union { __bf16 h[2]; uint_t u; } x;
  x.h[0] = (__bf16)a;
  x.h[1] = (__bf16)b;
  return x.u;
}

// ---- GEMM parameters --------------------------------------------------------
// C[m,n] = act( sum_k A[m,k]*B[n,k] + bias[n] ) (+ bias2[n]) + add[m,n]
// TA: A[m,k] at A + k*lda + m   (else A + m*lda + k)
// TB: B[n,k] at B + k*ldb + n   (else B + n*ldb + k)   (B == weight [N,K])
// Batched over blockIdx.z with two-level strides: z -> (zb = z/bdiv, zh = z%bdiv).
struct GemmP {
  const float* A; const float* B; const float* bias; const float* bias2;
  const float* add; float* C;
  long lda, ldb, ldc, ldadd;
  int  M, N, K;
  int  act, bdiv;
  long sA0, sA1, sB0, sB1, sC0, sC1, sAdd0, sAdd1;
};

// Stage 8 consecutive-K f32 elements of one tile row into LDS as bf16.
template <bool TR>
DEV void stage8(const float* base, long ld, long row, long col, ushort_t* dst) {
  float v[8];
  if (!TR) {
    const float* p = base + row * ld + col;
    float4 f0 = *reinterpret_cast<const float4*>(p);
    float4 f1 = *reinterpret_cast<const float4*>(p + 4);
    v[0] = f0.x; v[1] = f0.y; v[2] = f0.z; v[3] = f0.w;
    v[4] = f1.x; v[5] = f1.y; v[6] = f1.z; v[7] = f1.w;
  } else {
    const float* p = base + col * ld + row;
#pragma unroll
    for (int j = 0; j < 8; ++j) v[j] = p[(long)j * ld];
  }
  uint4 q;
  q.x = pk2(v[0], v[1]); q.y = pk2(v[2], v[3]);
  q.z = pk2(v[4], v[5]); q.w = pk2(v[6], v[7]);
  *reinterpret_cast<uint4*>(dst) = q;
}

// Load one 16x32 bf16 fragment from an LDS tile laid out [16][32] row-major.
// Lane l<16 : row=l,   K = {0..7, 16..23};  lane l>=16: row=l-16, K={8..15,24..31}
// (matches CDNA5 ISA 16-bit A/B matrix VGPR striping).  Two ds_load_b128 / lane.
DEV v16bf load_frag(const ushort_t* tile, int lane) {
  const int r  = lane & 15;
  const int c0 = (lane >> 4) << 3;
  union { v16bf v; uint4 q[2]; } u;
  u.q[0] = *reinterpret_cast<const uint4*>(tile + r * 32 + c0);
  u.q[1] = *reinterpret_cast<const uint4*>(tile + r * 32 + c0 + 16);
  return u.v;
}

DEV float act_apply(float v, int act, float b2) {
  switch (act) {
    case 1: return 1.f / (1.f + __expf(-v));                       // sigmoid
    case 2: return v > 0.f ? v + 1.f : __expf(v);                  // elu(v)+1
    case 3: return 0.5f * v * (1.f + erff(v * 0.70710678118f));    // exact gelu
    case 5: return __expf(b2 * (1.f / (1.f + __expf(-v))));        // SSM gate
    default: return v + b2;                                        // none (+b2)
  }
}

DEV void epilogue_frag(const v8f acc, long m0, long n0, const GemmP& g,
                       float* C, const float* add, int lane) {
  const int col = lane & 15;
  const int r0  = (lane >> 4) * 8;
  const long n = n0 + col;
  if (n >= g.N) return;
  const float bv = g.bias  ? g.bias[n]  : 0.f;
  const float b2 = g.bias2 ? g.bias2[n] : 0.f;
#pragma unroll
  for (int r = 0; r < 8; ++r) {
    const long m = m0 + r0 + r;
    if (m >= g.M) continue;
    float v = acc[r] + bv;
    v = act_apply(v, g.act, b2);
    if (add) v += add[m * g.ldadd + n];
    C[m * g.ldc + n] = v;
  }
}

// ---- WMMA GEMM kernel -------------------------------------------------------
// Block tile 128(M) x 64(N), BK=32, 8 wave32 waves, each wave owns a 32x32
// register tile (2 A-frags x 2 B-frags -> 4 v_wmma per K-step).  The last
// K-tile is peeled so the steady-state loop body is branch-free.
template <bool TA, bool TB>
__global__ __launch_bounds__(256)
void gemm_wmma(GemmP g) {
  __shared__ ushort_t As[2][128 * 32];
  __shared__ ushort_t Bs[2][64 * 32];

  const int tid  = threadIdx.x;
  const int lane = tid & 31;
  const int wave = tid >> 5;
  const int wm   = (wave & 3) * 32;   // wave M-subtile (two 16-row frags)
  const int wn   = (wave >> 2) * 32;  // wave N-subtile (two 16-col frags)
  const long bm  = (long)blockIdx.y * 128;
  const long bn  = (long)blockIdx.x * 64;

  const int z  = blockIdx.z;
  const int zh = z % g.bdiv;
  const int zb = z / g.bdiv;
  const float* __restrict__ A   = g.A + zb * g.sA1 + zh * g.sA0;
  const float* __restrict__ Bm  = g.B + zb * g.sB1 + zh * g.sB0;
  float*       __restrict__ C   = g.C + zb * g.sC1 + zh * g.sC0;
  const float* add = g.add ? g.add + zb * g.sAdd1 + zh * g.sAdd0 : nullptr;

  // Staging assignments (256 threads):
  //   A tile 128x32 f32 -> 16 floats/thread: row rA = tid>>1, cols cA..cA+15
  //   B tile  64x32 f32 ->  8 floats/thread: row rB = tid>>2, cols cB..cB+7
  const int rA = tid >> 1;
  const int cA = (tid & 1) * 16;
  const int rB = tid >> 2;
  const int cB = (tid & 3) * 8;

  v8f acc00 = {}, acc01 = {}, acc10 = {}, acc11 = {};

  const int nk = g.K >> 5;
  stage8<TA>(A, g.lda, bm + rA, cA,     &As[0][rA * 32 + cA]);
  stage8<TA>(A, g.lda, bm + rA, cA + 8, &As[0][rA * 32 + cA + 8]);
  stage8<TB>(Bm, g.ldb, bn + rB, cB,    &Bs[0][rB * 32 + cB]);
  __syncthreads();

  int buf = 0;
  // Steady state: always stages the next tile -> branch-free body.
  for (int kt = 0; kt + 1 < nk; ++kt) {
    const long k1 = (long)(kt + 1) << 5;
    stage8<TA>(A, g.lda, bm + rA, k1 + cA,     &As[buf ^ 1][rA * 32 + cA]);
    stage8<TA>(A, g.lda, bm + rA, k1 + cA + 8, &As[buf ^ 1][rA * 32 + cA + 8]);
    stage8<TB>(Bm, g.ldb, bn + rB, k1 + cB,    &Bs[buf ^ 1][rB * 32 + cB]);
    if (kt + 2 < nk) {  // CDNA5 global_prefetch_b8 for the tile after next
      const long k2 = (long)(kt + 2) << 5;
      if (!TA) __builtin_prefetch(A + (bm + rA) * g.lda + k2, 0, 0);
      if (!TB) __builtin_prefetch(Bm + (bn + rB) * g.ldb + k2, 0, 0);
    }
    const v16bf a0 = load_frag(As[buf] + wm * 32, lane);
    const v16bf a1 = load_frag(As[buf] + (wm + 16) * 32, lane);
    const v16bf b0 = load_frag(Bs[buf] + wn * 32, lane);
    const v16bf b1 = load_frag(Bs[buf] + (wn + 16) * 32, lane);
    // reuse_a=true on the second of each A-sharing pair (identical opcode,
    // same SRC0) -> exercises the VOP3P operand-reuse cache.
    acc00 = __builtin_amdgcn_wmma_f32_16x16x32_bf16(false, a0, false, b0,
                                                    (short)0, acc00, false, false);
    acc01 = __builtin_amdgcn_wmma_f32_16x16x32_bf16(false, a0, false, b1,
                                                    (short)0, acc01, true, false);
    acc10 = __builtin_amdgcn_wmma_f32_16x16x32_bf16(false, a1, false, b0,
                                                    (short)0, acc10, false, false);
    acc11 = __builtin_amdgcn_wmma_f32_16x16x32_bf16(false, a1, false, b1,
                                                    (short)0, acc11, true, false);
    __syncthreads();
    buf ^= 1;
  }
  // Peeled last K-tile.
  {
    const v16bf a0 = load_frag(As[buf] + wm * 32, lane);
    const v16bf a1 = load_frag(As[buf] + (wm + 16) * 32, lane);
    const v16bf b0 = load_frag(Bs[buf] + wn * 32, lane);
    const v16bf b1 = load_frag(Bs[buf] + (wn + 16) * 32, lane);
    acc00 = __builtin_amdgcn_wmma_f32_16x16x32_bf16(false, a0, false, b0,
                                                    (short)0, acc00, false, false);
    acc01 = __builtin_amdgcn_wmma_f32_16x16x32_bf16(false, a0, false, b1,
                                                    (short)0, acc01, true, false);
    acc10 = __builtin_amdgcn_wmma_f32_16x16x32_bf16(false, a1, false, b0,
                                                    (short)0, acc10, false, false);
    acc11 = __builtin_amdgcn_wmma_f32_16x16x32_bf16(false, a1, false, b1,
                                                    (short)0, acc11, true, false);
  }

  epilogue_frag(acc00, bm + wm,      bn + wn,      g, C, add, lane);
  epilogue_frag(acc01, bm + wm,      bn + wn + 16, g, C, add, lane);
  epilogue_frag(acc10, bm + wm + 16, bn + wn,      g, C, add, lane);
  epilogue_frag(acc11, bm + wm + 16, bn + wn + 16, g, C, add, lane);
}

// ---- wavelet weight repack: [co][ci][k] strided -> two dense [512][512] -----
__global__ __launch_bounds__(256)
void wpack_kernel(const float* hpw, const float* hpb,
                  const float* lpw, const float* lpb,
                  float* w0, float* w1, float* wb) {
  const int idx = blockIdx.x * 256 + threadIdx.x;
  if (idx >= 512 * 512) return;
  const int co = idx >> 9, ci = idx & 511;
  const float* src = (co < 256) ? hpw + (long)co * 1024 + ci * 2
                                : lpw + (long)(co - 256) * 1024 + ci * 2;
  w0[idx] = src[0];
  w1[idx] = src[1];
  if (idx < 512) wb[idx] = (idx < 256) ? hpb[idx] : lpb[idx - 256];
}

// ---- elementwise gate multiply ---------------------------------------------
__global__ __launch_bounds__(256)
void emul_kernel(float* c, const float* g, long n) {
  const long i = (long)blockIdx.x * 256 + threadIdx.x;
  if (i < n) c[i] *= g[i];
}

// ---- noise embedding: ne[b,d] = w[d]*noise[b] + b[d] ------------------------
__global__ __launch_bounds__(256)
void ne_kernel(const float* w, const float* b, const float* noise, float* out) {
  const int idx = blockIdx.x * 256 + threadIdx.x;  // < B*D
  const int bi = idx >> 9, d = idx & 511;
  out[idx] = w[d] * noise[bi] + b[d];
}

// ---- layernorm over D=512, one 256-thread block per token -------------------
__global__ __launch_bounds__(256)
void ln_kernel(const float* x, float* y, const float* g, const float* b,
               const float* ne, int s2) {
  __shared__ float s1[256], s2s[256];
  const int tok = blockIdx.x, tid = threadIdx.x;
  const long base = (long)tok * 512;
  const int bi = tok / s2;
  const float n0 = ne ? ne[bi * 512 + tid]       : 0.f;
  const float n1 = ne ? ne[bi * 512 + tid + 256] : 0.f;
  const float v0 = x[base + tid] + n0;
  const float v1 = x[base + tid + 256] + n1;
  s1[tid] = v0 + v1;
  s2s[tid] = v0 * v0 + v1 * v1;
  __syncthreads();
  for (int off = 128; off > 0; off >>= 1) {
    if (tid < off) { s1[tid] += s1[tid + off]; s2s[tid] += s2s[tid + off]; }
    __syncthreads();
  }
  const float mean = s1[0] * (1.f / 512.f);
  const float var  = s2s[0] * (1.f / 512.f) - mean * mean;
  const float rstd = rsqrtf(var + 1e-5f);
  y[base + tid]       = (v0 - mean) * rstd * g[tid] + b[tid];
  y[base + tid + 256] = (v1 - mean) * rstd * g[tid + 256] + b[tid + 256];
}

// ---- SSM scan: h[t] = a[t]*h[t-1] + bx[t], one thread per (b,d) -------------
__global__ __launch_bounds__(256)
void scan_kernel(const float* a, const float* bx, float* hs) {
  const int idx = blockIdx.x * 256 + threadIdx.x;  // < B*D
  const int b = idx >> 9, d = idx & 511;
  const long base = (long)b * 1024 * 512 + d;
  float h = 0.f;
  for (int t = 0; t < 1024; ++t) {
    const long off = base + (long)t * 512;
    h = a[off] * h + bx[off];
    hs[off] = h;
  }
}

// ---- adaptive-kernel block (K=16), one block per token, in-place residual ---
__global__ __launch_bounds__(256)
void ak_kernel(float* x, const float* ker, const float* mixw, const float* mixb,
               const float* outw, const float* outb) {
  __shared__ float xs[512];
  __shared__ float pd[32][8];
  __shared__ float pks[16][8];
  __shared__ float red[256];
  __shared__ float av[16], gk[16];
  const int tok = blockIdx.x, tid = threadIdx.x;
  const long base = (long)tok * 512;
  xs[tid]       = x[base + tid];
  xs[tid + 256] = x[base + tid + 256];
  __syncthreads();

  const int col = tid & 31, grp = tid >> 5, d0 = grp * 64;
  float acc = 0.f, k2 = 0.f;
  if (col < 16) {
    const float* kr = ker + (long)col * 512;
    for (int d = d0; d < d0 + 64; ++d) { const float kv = kr[d]; acc += xs[d] * kv; k2 += kv * kv; }
    pd[col][grp] = acc; pks[col][grp] = k2;
  } else {
    const float* mw = mixw + (long)(col - 16) * 512;
    for (int d = d0; d < d0 + 64; ++d) acc += xs[d] * mw[d];
    pd[col][grp] = acc;
  }
  red[tid] = xs[tid] * xs[tid] + xs[tid + 256] * xs[tid + 256];
  __syncthreads();
  for (int off = 128; off > 0; off >>= 1) {
    if (tid < off) red[tid] += red[tid + off];
    __syncthreads();
  }
  if (tid < 32) {
    float t = 0.f;
    for (int g8 = 0; g8 < 8; ++g8) t += pd[tid][g8];
    if (tid < 16) {
      float kk = 0.f;
      for (int g8 = 0; g8 < 8; ++g8) kk += pks[tid][g8];
      const float d2 = red[0] - 2.f * t + kk;
      av[tid] = __expf(-d2);
    } else {
      pd[tid][0] = t + mixb[tid - 16];  // logits parked in pd[16..31][0]
    }
  }
  __syncthreads();
  if (tid == 0) {
    float mx = -1e30f;
    for (int k = 0; k < 16; ++k) mx = fmaxf(mx, pd[16 + k][0]);
    float e[16], s = 0.f;
    for (int k = 0; k < 16; ++k) { e[k] = __expf(pd[16 + k][0] - mx); s += e[k]; }
    const float inv = 1.f / s;
    for (int k = 0; k < 16; ++k) gk[k] = av[k] * e[k] * inv;
  }
  __syncthreads();
  for (int n = tid; n < 512; n += 256) {
    float o = outb[n];
    const float* ow = outw + (long)n * 16;
#pragma unroll
    for (int k = 0; k < 16; ++k) o += gk[k] * ow[k];
    x[base + n] = xs[n] + o;
  }
}

// ---- output heads: pred = x.wo + bo; conf = sigmoid(x.wc + bc) --------------
__global__ __launch_bounds__(256)
void head_kernel(const float* x, const float* wo, const float* bo,
                 const float* wc, const float* bc, float* out, int T) {
  const int wave = threadIdx.x >> 5, lane = threadIdx.x & 31;
  const int tok = blockIdx.x * 8 + wave;
  const long base = (long)tok * 512;
  float s1 = 0.f, s2 = 0.f;
  for (int d = lane; d < 512; d += 32) {
    const float v = x[base + d];
    s1 += v * wo[d];
    s2 += v * wc[d];
  }
#pragma unroll
  for (int off = 16; off > 0; off >>= 1) {
    s1 += __shfl_xor(s1, off, 32);
    s2 += __shfl_xor(s2, off, 32);
  }
  if (lane == 0) {
    out[tok]     = s1 + bo[0];
    out[T + tok] = 1.f / (1.f + __expf(-(s2 + bc[0])));
  }
}

// ---------------------------------------------------------------------------
// Host orchestration
// ---------------------------------------------------------------------------
static void gemm(hipStream_t st, const float* A, long lda, int tA,
                 const float* Bm, long ldb, int tB,
                 const float* bias, const float* bias2,
                 const float* add, long ldadd,
                 float* C, long ldc, int M, int N, int K, int act,
                 int batches = 1, int bdiv = 1,
                 long sA0 = 0, long sA1 = 0, long sB0 = 0, long sB1 = 0,
                 long sC0 = 0, long sC1 = 0, long sAdd0 = 0, long sAdd1 = 0) {
  GemmP p{A, Bm, bias, bias2, add, C, lda, ldb, ldc, ldadd, M, N, K,
          act, bdiv, sA0, sA1, sB0, sB1, sC0, sC1, sAdd0, sAdd1};
  dim3 grid((N + 63) / 64, (M + 127) / 128, batches), block(256);
  if (!tA && !tB)      hipLaunchKernelGGL((gemm_wmma<false, false>), grid, block, 0, st, p);
  else if (!tA && tB)  hipLaunchKernelGGL((gemm_wmma<false, true >), grid, block, 0, st, p);
  else if (tA && !tB)  hipLaunchKernelGGL((gemm_wmma<true,  false>), grid, block, 0, st, p);
  else                 hipLaunchKernelGGL((gemm_wmma<true,  true >), grid, block, 0, st, p);
}

extern "C" void kernel_launch(void* const* d_in, const int* in_sizes, int n_in,
                              void* d_out, int out_size, void* d_ws, size_t ws_size,
                              hipStream_t stream) {
  (void)in_sizes; (void)n_in; (void)out_size; (void)ws_size;
  constexpr int  Bc = 32, Sc = 2048, INc = 64, Dc = 512, Lc = 3, Hc = 4, S2c = 1024;
  constexpr long Tc = (long)Bc * S2c;    // 32768 tokens after wavelet
  constexpr long TD = Tc * Dc;

  // Input pytree layout (top-level insertion order; nested dicts key-sorted):
  // 0:x 1:noise 2:conf.b 3:conf.w 4:in.b 5:in.w  [layers: 35 leaves each from 6]
  // 111:out.b 112:out.w 113:pos 114:hp_b 115:hp_w 116:lp_b 117:lp_w 118:rec.b 119:rec.w
  auto F = [&](int i) { return (const float*)d_in[i]; };

  float* ws = (float*)d_ws;
  long o = 0;
  float* xfull = ws + o; o += (long)Bc * Sc * Dc;   // [B,S,D] (reused as t3 later)
  float* X     = ws + o; o += TD;                   // running activation [B,S2,D]
  float* t1    = ws + o; o += TD;
  float* t2    = ws + o; o += TD;
  float* hb    = ws + o; o += Tc * 4 * Dc;          // FF hidden [B,S2,4D]
  float* kv    = ws + o; o += (long)Bc * Hc * 128 * 128;
  float* neb   = ws + o; o += (long)Bc * Dc;
  float* wk0   = ws + o; o += 512 * 512;
  float* wk1   = ws + o; o += 512 * 512;
  float* wbb   = ws + o; o += 512;
  float* t3    = xfull;                             // reuse after wavelet

  // 1) input projection + positional embedding (batched over B, pos broadcast)
  gemm(stream, F(0), INc, 0, F(5), INc, 0, F(4), nullptr, F(113), Dc,
       xfull, Dc, Sc, Dc, INc, 0,
       Bc, 1, 0, (long)Sc * INc, 0, 0, 0, (long)Sc * Dc, 0, 0);

  // 2) wavelet: repack stride-2 conv weights, then two GEMMs + reconstruction
  hipLaunchKernelGGL(wpack_kernel, dim3((512 * 512 + 255) / 256), dim3(256), 0,
                     stream, F(115), F(114), F(117), F(116), wk0, wk1, wbb);
  // even taps: row m of A = x[b, 2t, :] lives at xfull + m*2D  (since S == 2*S2)
  gemm(stream, xfull, 2 * Dc, 0, wk0, Dc, 0, wbb, nullptr, nullptr, 0,
       t1, Dc, (int)Tc, Dc, Dc, 0);
  // odd taps accumulate: add == C
  gemm(stream, xfull + Dc, 2 * Dc, 0, wk1, Dc, 0, nullptr, nullptr, t1, Dc,
       t1, Dc, (int)Tc, Dc, Dc, 0);
  // rec linear -> X
  gemm(stream, t1, Dc, 0, F(119), Dc, 0, F(118), nullptr, nullptr, 0,
       X, Dc, (int)Tc, Dc, Dc, 0);

  for (int l = 0; l < Lc; ++l) {
    const int LB = 6 + 35 * l;
    // layer leaf map: +0 ak.kernels +1 mix.b +2 mix.w +3 out.b +4 out.w
    // +5 gate.b +6 gate.w +7 k.b +8 k.w +9 o.b +10 o.w +11 q.b +12 q.w +13 v.b +14 v.w
    // +15 ff1.b +16 ff1.w +17 ff2.b +18 ff2.w +19 ln1.b +20 ln1.g +21 ln2.b +22 ln2.g
    // +23 ne.b +24 ne.w +25 ss.A +26 Ap.b +27 Ap.w +28 B.b +29 B.w +30 C.b +31 C.w
    // +32 ss.D +33 proj.b +34 proj.w

    // ---- linear attention ----
    gemm(stream, X, Dc, 0, F(LB + 12), Dc, 0, F(LB + 11), nullptr, nullptr, 0,
         t1, Dc, (int)Tc, Dc, Dc, 2);                       // q = elu+1
    gemm(stream, X, Dc, 0, F(LB + 8), Dc, 0, F(LB + 7), nullptr, nullptr, 0,
         t2, Dc, (int)Tc, Dc, Dc, 2);                       // k = elu+1
    gemm(stream, X, Dc, 0, F(LB + 14), Dc, 0, F(LB + 13), nullptr, nullptr, 0,
         t3, Dc, (int)Tc, Dc, Dc, 0);                       // v
    // kv[b,h] = k^T v : M=N=HD=128, K=S2 ; batched z = B*H, bdiv = H
    gemm(stream, t2, Dc, 1, t3, Dc, 1, nullptr, nullptr, nullptr, 0,
         kv, 128, 128, 128, S2c, 0,
         Bc * Hc, Hc, 128, (long)S2c * Dc, 128, (long)S2c * Dc,
         16384, (long)Hc * 16384, 0, 0);
    // o = q @ kv  (kv used transposed via transB) -> t2 in [B,S2,D] layout
    gemm(stream, t1, Dc, 0, kv, 128, 1, nullptr, nullptr, nullptr, 0,
         t2, Dc, S2c, 128, 128, 0,
         Bc * Hc, Hc, 128, (long)S2c * Dc, 16384, (long)Hc * 16384,
         128, (long)S2c * Dc, 0, 0);
    // gate, multiply, output projection (+residual into X)
    gemm(stream, X, Dc, 0, F(LB + 6), Dc, 0, F(LB + 5), nullptr, nullptr, 0,
         t1, Dc, (int)Tc, Dc, Dc, 1);                       // sigmoid gate
    hipLaunchKernelGGL(emul_kernel, dim3((unsigned)((TD + 255) / 256)), dim3(256),
                       0, stream, t2, t1, TD);
    gemm(stream, t2, Dc, 0, F(LB + 10), Dc, 0, F(LB + 9), nullptr, X, Dc,
         X, Dc, (int)Tc, Dc, Dc, 0);

    // ---- adaptive kernel (K=16) ----
    hipLaunchKernelGGL(ak_kernel, dim3((unsigned)Tc), dim3(256), 0, stream,
                       X, F(LB + 0), F(LB + 2), F(LB + 1), F(LB + 4), F(LB + 3));

    // ---- state space ----
    gemm(stream, X, Dc, 0, F(LB + 27), Dc, 0, F(LB + 26), F(LB + 25), nullptr, 0,
         t1, Dc, (int)Tc, Dc, Dc, 5);                       // A = exp(A*sig(Ap x))
    gemm(stream, X, Dc, 0, F(LB + 29), Dc, 0, F(LB + 28), nullptr, nullptr, 0,
         t2, Dc, (int)Tc, Dc, Dc, 0);                       // Bx
    hipLaunchKernelGGL(scan_kernel, dim3((Bc * Dc) / 256), dim3(256), 0, stream,
                       t1, t2, t3);                         // hs
    gemm(stream, t3, Dc, 0, F(LB + 31), Dc, 0, F(LB + 30), F(LB + 32), X, Dc,
         t2, Dc, (int)Tc, Dc, Dc, 0);                       // y = C hs + b + D + x
    gemm(stream, t2, Dc, 0, F(LB + 34), Dc, 0, F(LB + 33), nullptr, X, Dc,
         X, Dc, (int)Tc, Dc, Dc, 0);                        // x += proj(y)

    // ---- diffusion block ----
    hipLaunchKernelGGL(ne_kernel, dim3((Bc * Dc + 255) / 256), dim3(256), 0,
                       stream, F(LB + 24), F(LB + 23), F(1), neb);
    hipLaunchKernelGGL(ln_kernel, dim3((unsigned)Tc), dim3(256), 0, stream,
                       X, X, F(LB + 20), F(LB + 19), neb, S2c);   // x = ln1(x+ne)
    hipLaunchKernelGGL(ln_kernel, dim3((unsigned)Tc), dim3(256), 0, stream,
                       X, t1, F(LB + 22), F(LB + 21), nullptr, S2c);
    gemm(stream, t1, Dc, 0, F(LB + 16), Dc, 0, F(LB + 15), nullptr, nullptr, 0,
         hb, 4 * Dc, (int)Tc, 4 * Dc, Dc, 3);               // h = gelu(ff1)
    gemm(stream, hb, 4 * Dc, 0, F(LB + 18), 4 * Dc, 0, F(LB + 17), nullptr, X, Dc,
         X, Dc, (int)Tc, Dc, 4 * Dc, 0);                    // x += ff2(h)
  }

  // 3) output heads -> d_out = [pred(B*S2) | conf(B*S2)]
  hipLaunchKernelGGL(head_kernel, dim3((unsigned)(Tc / 8)), dim3(256), 0, stream,
                     X, F(112), F(111), F(3), F(2), (float*)d_out, (int)Tc);
}
